// MyGRU_19069654794505
// MI455X (gfx1250) — matrix operand order repaired and checked by reading, not compiled, not run on hardware
//
#include <hip/hip_runtime.h>
#include <hip/hip_bf16.h>

// GRU persistent kernel for MI455X (gfx1250, wave32).
// B=64, L=2048, H=512, IN=1.
//
// 32 workgroups x 128 threads (4 waves). Each WG owns 16 hidden columns
// (all 3 gates -> 48 N-cols); its bf16 W_hh slice (48KB) lives in LDS in
// WMMA B-fragment layout. Each wave owns a 16x16 (batch x col) patch of the
// recurrent state: fp32 residual + fp32 st accumulator stay in VGPRs for all
// 2048 steps. Per step: 3 gate GEMM tiles via v_wmma_f32_16x16x32_bf16
// (K=512 -> 16 WMMAs per gate), gate nonlinearities in fp32, st written to
// out, new h written as bf16 into a ping-pong global buffer in A-fragment
// layout, then a device-wide generation barrier.

#define B_  64
#define L_  2048
#define H_  512
#define NWG 32
#define TPB 128

typedef __attribute__((ext_vector_type(16))) __bf16 v16bf;
typedef __attribute__((ext_vector_type(8)))  float  v8f;

union Frag32 { uint4 q[2]; v16bf v; };

__device__ __forceinline__ unsigned short f2bf(float f) {
  union { float f; unsigned u; } c; c.f = f;
  unsigned r = c.u + 0x7FFFu + ((c.u >> 16) & 1u);   // round-to-nearest-even
  return (unsigned short)(r >> 16);
}

__device__ __forceinline__ float fsigmoid(float x) {
  return __builtin_amdgcn_rcpf(1.0f + __expf(-x));
}
__device__ __forceinline__ float ftanhf(float x) {
  return 1.0f - 2.0f * __builtin_amdgcn_rcpf(__expf(2.0f * x) + 1.0f);
}

// ---- init: barrier counter = 0; hA0 = bf16(state) in WMMA A-fragment layout.
// A-frag (16-bit, 16x32): element (M,K): lane = ((K>>3)&1)*16 + M,
//                         e = (K>>4)*8 + (K&7); frag index [mtile][ktile].
__global__ void gru_init(const float* __restrict__ state,
                         unsigned short* __restrict__ hA0,
                         int* __restrict__ barcnt) {
  int i = blockIdx.x * blockDim.x + threadIdx.x;
  if (i == 0) *barcnt = 0;
  if (i < B_ * H_) {
    int Mg = i / H_, Kg = i % H_;
    int mt = Mg >> 4, kt = Kg >> 5, M = Mg & 15, K = Kg & 31;
    int ln = (((K >> 3) & 1) << 4) + M;
    int e  = ((K >> 4) << 3) + (K & 7);
    hA0[(((mt * 16 + kt) * 32 + ln) << 4) + e] = f2bf(state[i]);
  }
}

__global__ void __launch_bounds__(TPB, 1) gru_persist(
    const float* __restrict__ x,      // [B, L]
    const float* __restrict__ state,  // [B, H]
    const float* __restrict__ W_ih,   // [3H, 1]
    const float* __restrict__ W_hh,   // [3H, H]
    const float* __restrict__ b_ih,   // [3H]
    const float* __restrict__ b_hh,   // [3H]
    const float* __restrict__ Aptr,   // [1]
    float* __restrict__ out,          // [B, L, H]
    float* __restrict__ fin,          // [B, H]
    unsigned short* __restrict__ hA0,
    unsigned short* __restrict__ hA1,
    int* __restrict__ barcnt) {
  __shared__ __align__(16) unsigned short wlds[3 * 16 * 32 * 16]; // 48KB
  __shared__ __align__(16) float xsh[B_];

  const int wg    = blockIdx.x;      // 0..31  -> hidden cols [wg*16, wg*16+16)
  const int tid   = threadIdx.x;
  const int lane  = tid & 31;
  const int mtile = tid >> 5;        // 0..3   -> batch rows [mtile*16, +16)

  // ---- Stage W_hh slice into LDS as bf16 B-fragments.
  // B-frag (16-bit, 32x16): element (k,n): lane = n + 16*((k>>3)&1), e = (k>>4)*8 + (k&7).
  // LDS layout: [gate 0..2][ktile 0..15][lane 0..31][e 0..15].
  for (int f = tid; f < 3 * 16 * 32 * 16; f += TPB) {
    int e  = f & 15;
    int ln = (f >> 4) & 31;
    int kt = (f >> 9) & 15;
    int nt = f >> 13;                                  // gate
    int col = nt * H_ + wg * 16 + (ln & 15);           // row of W_hh
    int k   = kt * 32 + ((ln >> 4) << 3) + (e & 7) + ((e >> 3) << 4);
    wlds[f] = f2bf(W_hh[col * H_ + k]);
  }

  // ---- Per-lane gate constants (N = lane&15 fixed within the wave's patch).
  const int ch = wg * 16 + (lane & 15);                // hidden column
  const float wih_r = W_ih[ch];
  const float wih_z = W_ih[H_ + ch];
  const float wih_n = W_ih[2 * H_ + ch];
  const float bs_r  = b_ih[ch] + b_hh[ch];
  const float bs_z  = b_ih[H_ + ch] + b_hh[H_ + ch];
  const float bi_n  = b_ih[2 * H_ + ch];
  const float bh_n  = b_hh[2 * H_ + ch];
  const float Aval  = Aptr[0];

  // ---- Persistent fp32 state: C-layout element r -> (M = 8*(lane>>4)+r).
  const int Mg0 = 16 * mtile + 8 * (lane >> 4);
  float resid[8], st[8];
#pragma unroll
  for (int r = 0; r < 8; ++r) {
    float h0 = state[(Mg0 + r) * H_ + ch];
    resid[r] = h0; st[r] = h0;
  }

  // ---- Producer scatter coords for writing h (A-fragment layout).
  const int ktP     = wg >> 1;
  const int lnPbase = (((lane >> 3) & 1) << 4) + 8 * (lane >> 4);
  const int eP      = ((wg & 1) << 3) + (lane & 7);

  __syncthreads();   // wlds ready

  for (int t = 0; t < L_; ++t) {
    const unsigned short* hAr = (t & 1) ? hA1 : hA0;
    unsigned short*       hAw = (t & 1) ? hA0 : hA1;

    if (tid < B_) xsh[tid] = x[tid * L_ + t];
    __syncthreads();

    v8f accR = {}, accZ = {}, accN = {};
#pragma unroll 4
    for (int kt = 0; kt < 16; ++kt) {
      Frag32 a, br, bz, bn;
      const uint4* ap = (const uint4*)(hAr + (((mtile * 16 + kt) * 32 + lane) << 4));
      a.q[0] = ap[0]; a.q[1] = ap[1];
      const uint4* bp0 = (const uint4*)(wlds + (((0 * 16 + kt) * 32 + lane) << 4));
      br.q[0] = bp0[0]; br.q[1] = bp0[1];
      const uint4* bp1 = (const uint4*)(wlds + (((1 * 16 + kt) * 32 + lane) << 4));
      bz.q[0] = bp1[0]; bz.q[1] = bp1[1];
      const uint4* bp2 = (const uint4*)(wlds + (((2 * 16 + kt) * 32 + lane) << 4));
      bn.q[0] = bp2[0]; bn.q[1] = bp2[1];
      accR = __builtin_amdgcn_wmma_f32_16x16x32_bf16(false, a.v, false, br.v, (short)0, accR, false, false);
      accZ = __builtin_amdgcn_wmma_f32_16x16x32_bf16(false, a.v, false, bz.v, (short)0, accZ, false, false);
      accN = __builtin_amdgcn_wmma_f32_16x16x32_bf16(false, a.v, false, bn.v, (short)0, accN, false, false);
    }

#pragma unroll
    for (int r = 0; r < 8; ++r) {
      float xv = xsh[Mg0 + r];
      float rg = fsigmoid(xv * wih_r + bs_r + accR[r]);
      float zg = fsigmoid(xv * wih_z + bs_z + accZ[r]);
      float ng = ftanhf(xv * wih_n + bi_n + rg * (accN[r] + bh_n));
      float hn = (1.0f - zg) * ng + zg * resid[r];
      resid[r] = hn;
      float s = st[r] + Aval * hn;
      st[r] = s;
      out[((size_t)(Mg0 + r) * L_ + t) * H_ + ch] = s;
      hAw[(((mtile * 16 + ktP) * 32 + (lnPbase + r)) << 4) + eP] = f2bf(hn);
    }

    // ---- Device-wide generation barrier (one per step; ping-pong buffers).
    __threadfence();
    __syncthreads();
    if (tid == 0) {
      __hip_atomic_fetch_add(barcnt, 1, __ATOMIC_RELEASE, __HIP_MEMORY_SCOPE_AGENT);
      const int target = (t + 1) * NWG;
      while (__hip_atomic_load(barcnt, __ATOMIC_ACQUIRE, __HIP_MEMORY_SCOPE_AGENT) < target)
        __builtin_amdgcn_s_sleep(2);
    }
    __syncthreads();
    __threadfence();   // make remote hA stores visible before next-step reads
  }

#pragma unroll
  for (int r = 0; r < 8; ++r)
    fin[(size_t)(Mg0 + r) * H_ + ch] = st[r];
}

extern "C" void kernel_launch(void* const* d_in, const int* in_sizes, int n_in,
                              void* d_out, int out_size, void* d_ws, size_t ws_size,
                              hipStream_t stream) {
  const float* x     = (const float*)d_in[0];
  const float* state = (const float*)d_in[1];
  const float* W_ih  = (const float*)d_in[2];
  const float* W_hh  = (const float*)d_in[3];
  const float* b_ih  = (const float*)d_in[4];
  const float* b_hh  = (const float*)d_in[5];
  const float* A     = (const float*)d_in[6];

  float* out = (float*)d_out;
  float* fin = out + (size_t)B_ * L_ * H_;

  // Workspace layout: [0,128)=barrier counter, then two 64KB bf16 h buffers.
  char* ws = (char*)d_ws;
  int* barcnt = (int*)ws;
  unsigned short* hA0 = (unsigned short*)(ws + 128);
  unsigned short* hA1 = (unsigned short*)(ws + 128 + B_ * H_ * 2);

  gru_init<<<(B_ * H_ + 255) / 256, 256, 0, stream>>>(state, hA0, barcnt);
  gru_persist<<<NWG, TPB, 0, stream>>>(x, state, W_ih, W_hh, b_ih, b_hh, A,
                                       out, fin, hA0, hA1, barcnt);
}
